// TiSASRec_9397388444270
// MI455X (gfx1250) — matrix operand order repaired and checked by reading, not compiled, not run on hardware
//
#include <hip/hip_runtime.h>
#include <hip/hip_bf16.h>
#include <math.h>

// TiSASRec forward for MI455X (gfx1250, wave32, WMMA + TDM).
// B=8, L=256, D=128, H=4, HD=32, NB=2. M = B*L = 2048 rows.
// Workspace layout (floats): seqs, q_in, Q, K, V, attn, hbuf -> 7 * 1MB = 7MB.

#define BB 8
#define LL 256
#define DD 128
#define HH 4
#define HD 32
#define MM (BB * LL)
#define NEGV (-4294967295.0f)     // -(2^32)+1
#define INVSQRT_HD 0.17677669529663687f
#define SQRT_D 11.313708498984761f

typedef __attribute__((ext_vector_type(16))) _Float16 v16h;
typedef __attribute__((ext_vector_type(8)))  float    v8f;
typedef __attribute__((ext_vector_type(4)))  unsigned int uint32x4;
typedef __attribute__((ext_vector_type(8)))  int      int32x8;
typedef __attribute__((ext_vector_type(4)))  int      int32x4;

// ---------------------------------------------------------------------------
// WMMA fragment helpers (ISA m/n lane mapping, contiguous k order, wave32)
// ---------------------------------------------------------------------------

// A-matrix 16x32 f16: lane<16 -> M=lane, K=0..15;
//                     lane>=16 -> M=lane-16, K=16..31 (contiguous).
__device__ __forceinline__ v16h frag_ld_a_h(const _Float16* base, int ld) {
  const int lane = threadIdx.x & 31;
  const _Float16* row = base + (size_t)(lane & 15) * ld + ((lane >> 4) << 4);
  v16h x;
#pragma unroll
  for (int i = 0; i < 16; ++i) x[i] = row[i];
  return x;
}

// A-fragment from an f32 source (LDS staging filled by the TDM).
__device__ __forceinline__ v16h frag_ld_a_f32(const float* base, int ld) {
  const int lane = threadIdx.x & 31;
  const float* row = base + (size_t)(lane & 15) * ld + ((lane >> 4) << 4);
  v16h x;
#pragma unroll
  for (int i = 0; i < 16; ++i) x[i] = (_Float16)row[i];
  return x;
}

// B-matrix 32x16 f16 loaded from B^T stored row-major (row n holds column n):
// lane<16 -> N=lane, K=0..15; lane>=16 -> N=lane-16, K=16..31 (contiguous).
__device__ __forceinline__ v16h frag_ld_b_h(const _Float16* baseT, int ld) {
  const int lane = threadIdx.x & 31;
  const _Float16* row = baseT + (size_t)(lane & 15) * ld + ((lane >> 4) << 4);
  v16h x;
#pragma unroll
  for (int i = 0; i < 16; ++i) x[i] = row[i];
  return x;
}

// Same as above but converting from an f32 source (global, L2-resident).
__device__ __forceinline__ v16h frag_ld_b_f32(const float* baseT, int ld) {
  const int lane = threadIdx.x & 31;
  const float* row = baseT + (size_t)(lane & 15) * ld + ((lane >> 4) << 4);
  v16h x;
#pragma unroll
  for (int i = 0; i < 16; ++i) x[i] = (_Float16)row[i];
  return x;
}

__device__ __forceinline__ v8f wmma16(v16h a, v16h b, v8f c) {
  // D = A(16x32 f16) * B(32x16 f16) + C(16x16 f32)
  return __builtin_amdgcn_wmma_f32_16x16x32_f16(false, a, false, b,
                                                (short)0, c, false, false);
}

// ---------------------------------------------------------------------------
// Elementwise / LN kernels
// ---------------------------------------------------------------------------

__global__ void embed_kernel(const float* __restrict__ item_emb,
                             const int* __restrict__ log_seqs,
                             float* __restrict__ seqs) {
  const int row = blockIdx.x, t = threadIdx.x;
  const int id = log_seqs[row];
  float v = item_emb[(size_t)id * DD + t] * SQRT_D;
  seqs[(size_t)row * DD + t] = (id == 0) ? 0.0f : v;
}

__global__ void ln_kernel(const float* x, const float* g, const float* b,
                          float* out) {
  __shared__ float red[DD];
  const int row = blockIdx.x, t = threadIdx.x;
  float v = x[(size_t)row * DD + t];
  red[t] = v;
  __syncthreads();
  for (int s = 64; s > 0; s >>= 1) { if (t < s) red[t] += red[t + s]; __syncthreads(); }
  const float mean = red[0] * (1.0f / DD);
  __syncthreads();
  const float d = v - mean;
  red[t] = d * d;
  __syncthreads();
  for (int s = 64; s > 0; s >>= 1) { if (t < s) red[t] += red[t + s]; __syncthreads(); }
  const float var = red[0] * (1.0f / DD);
  __syncthreads();
  out[(size_t)row * DD + t] = d * rsqrtf(var + 1e-8f) * g[t] + b[t];
}

__global__ void add_kernel(const float* __restrict__ a,
                           const float* __restrict__ b,
                           float* __restrict__ out, int n) {
  const int i = blockIdx.x * blockDim.x + threadIdx.x;
  if (i < n) out[i] = a[i] + b[i];
}

__global__ void mask_kernel(float* seqs, const int* __restrict__ log_seqs) {
  const int row = blockIdx.x, t = threadIdx.x;
  if (log_seqs[row] == 0) seqs[(size_t)row * DD + t] = 0.0f;
}

// ---------------------------------------------------------------------------
// WMMA GEMM: out[M,128] = act(A[M,128] @ W[128,128] + bias) (+ resid)
// 2 waves / workgroup, 32 M-rows per workgroup.
// A tile staged f32 in LDS by the Tensor Data Mover; W staged transposed f16.
// ---------------------------------------------------------------------------

template <bool RELU, bool RESID>
__global__ __launch_bounds__(64) void gemm128_kernel(
    const float* __restrict__ A, const float* __restrict__ W,
    const float* __restrict__ bias, const float* resid, float* out) {
  __shared__ float    a_f32[32 * 128];   // [m][k]  16KB, TDM target
  __shared__ _Float16 w_sh[128 * 128];   // [n][k]  32KB (W transposed)
  const int tid = threadIdx.x;
  const int mBlk = blockIdx.x * 32;
  const unsigned int nElem = 32u * 128u; // elements in the A tile

#if __has_builtin(__builtin_amdgcn_tensor_load_to_lds)
  // Wave 0 issues one TDM descriptor: 1-D tile of 4096 f32, global -> LDS.
  if ((tid >> 5) == 0) {
    const unsigned long long ga =
        (unsigned long long)(const void*)(A + (size_t)mBlk * DD);
    const unsigned int lds_off = (unsigned int)(uintptr_t)(void*)a_f32;
    uint32x4 g0;
    g0[0] = 1u;                                   // count=1, user descriptor
    g0[1] = lds_off;                              // lds_addr
    g0[2] = (unsigned int)ga;                     // global_addr[31:0]
    g0[3] = (unsigned int)((ga >> 32) & 0x01FFFFFFull) | (2u << 30); // type=2
    int32x8 g1;
    g1[0] = (int)(2u << 16);                      // data_size = 4B
    g1[1] = (int)(nElem << 16);                   // tensor_dim0[15:0]
    g1[2] = (int)((nElem >> 16) | (1u << 16));    // tensor_dim0 hi | tensor_dim1=1
    g1[3] = (int)(nElem << 16);                   // tile_dim0 = 4096
    g1[4] = 0;                                    // tile_dim1/2 unused
    g1[5] = (int)nElem;                           // tensor_dim0_stride
    g1[6] = 0;
    g1[7] = 0;
    int32x4 gz4 = {0, 0, 0, 0};
    int32x8 gz8 = {0, 0, 0, 0, 0, 0, 0, 0};
    __builtin_amdgcn_tensor_load_to_lds(g0, g1, gz4, gz4, gz8, 0);
    __builtin_amdgcn_s_wait_tensorcnt(0);
  }
#else
  for (int idx = tid; idx < 32 * 128; idx += 64)
    a_f32[idx] = A[(size_t)mBlk * DD + idx];
#endif

  for (int idx = tid; idx < 128 * 128; idx += 64) {
    const int nn = idx >> 7, k = idx & 127;
    w_sh[idx] = (_Float16)W[(size_t)k * DD + nn];
  }
  __syncthreads();

  const int lane = tid & 31, wv = tid >> 5;
  const int n15 = lane & 15, mo = (lane >> 4) << 3;
  const int mRow = mBlk + wv * 16;
#pragma unroll
  for (int nt = 0; nt < 8; ++nt) {
    v8f c = {0.f, 0.f, 0.f, 0.f, 0.f, 0.f, 0.f, 0.f};
#pragma unroll
    for (int kc = 0; kc < 4; ++kc) {
      v16h a  = frag_ld_a_f32(a_f32 + wv * 16 * 128 + kc * 32, 128);
      v16h bb = frag_ld_b_h(w_sh + nt * 16 * 128 + kc * 32, 128);
      c = wmma16(a, bb, c);
    }
    const int ncol = nt * 16 + n15;
    const float bv = bias[ncol];
#pragma unroll
    for (int r = 0; r < 8; ++r) {
      const int m = mRow + mo + r;
      float v = c[r] + bv;
      if (RELU) v = fmaxf(v, 0.0f);
      if (RESID) v += resid[(size_t)m * DD + ncol];
      out[(size_t)m * DD + ncol] = v;
    }
  }
}

// ---------------------------------------------------------------------------
// Attention: one wave per (b, h, 16-query tile).
//   scores = (QK^T + Q posK^T + tK.Q) / sqrt(HD), mask(pad_q | causal)
//   out    = A (V + posV) via WMMA  +  A.tV via VALU
// ---------------------------------------------------------------------------

__global__ __launch_bounds__(32) void attn_kernel(
    const float* __restrict__ Q, const float* __restrict__ K,
    const float* __restrict__ V, const float* __restrict__ posK,
    const float* __restrict__ posV, const float* __restrict__ timeK,
    const float* __restrict__ timeV, const int* __restrict__ tmx,
    const int* __restrict__ log_seqs, float* __restrict__ out) {
  __shared__ _Float16 q_h[16 * 32];     // Q tile f16 (A-fragment source)
  __shared__ float    q_f[16 * 32];     // Q tile f32 (tK VALU)
  __shared__ float    s_sc[16 * 256];   // scores / exp scratch
  __shared__ _Float16 p_h[16 * 256];    // probabilities f16
  __shared__ _Float16 vT[32 * 256];     // (V + posV)^T f16: [d][k]
  __shared__ int      tmr[16 * 256];    // time indices for this q-tile

  const int lane = threadIdx.x;
  const int qt = blockIdx.x & 15;          // L/16 = 16 query tiles
  const int h  = (blockIdx.x >> 4) & 3;    // H = 4
  const int b  = blockIdx.x >> 6;          // B = 8
  const int q0 = qt * 16;
  const int hd0 = h * HD;

  for (int idx = lane; idx < 16 * 32; idx += 32) {
    const int r = idx >> 5, d = idx & 31;
    const float v = Q[((size_t)(b * LL + q0 + r)) * DD + hd0 + d];
    q_f[idx] = v;
    q_h[idx] = (_Float16)v;
  }
  for (int idx = lane; idx < 256 * 32; idx += 32) {
    const int k = idx >> 5, d = idx & 31;
    const float v = V[((size_t)(b * LL + k)) * DD + hd0 + d] +
                    posV[(size_t)k * DD + hd0 + d];
    vT[d * 256 + k] = (_Float16)v;
  }
  for (int idx = lane; idx < 16 * 256; idx += 32) {
    const int r = idx >> 8, k = idx & 255;
    tmr[idx] = tmx[((size_t)b * LL + q0 + r) * LL + k];
  }
  __syncthreads();

  const v16h aq = frag_ld_a_h(q_h, 32);
  const int n  = lane & 15;
  const int mo = (lane >> 4) << 3;

  // ---- scores ----
  for (int kt = 0; kt < 16; ++kt) {
    const int k0 = kt * 16;
    const v16h bk = frag_ld_b_f32(K + ((size_t)(b * LL + k0)) * DD + hd0, DD);
    const v16h bp = frag_ld_b_f32(posK + (size_t)k0 * DD + hd0, DD);
    v8f c = {0.f, 0.f, 0.f, 0.f, 0.f, 0.f, 0.f, 0.f};
    c = wmma16(aq, bk, c);
    c = wmma16(aq, bp, c);
    const int kk = k0 + n;
#pragma unroll
    for (int r = 0; r < 8; ++r) {
      const int m = mo + r;
      const int q = q0 + m;
      const int tix = tmr[m * 256 + kk];
      const float* te = timeK + (size_t)tix * DD + hd0;
      const float* qr = q_f + m * 32;
      float acc = 0.0f;
#pragma unroll
      for (int d = 0; d < 32; ++d) acc = fmaf(te[d], qr[d], acc);
      const float sc = (c[r] + acc) * INVSQRT_HD;
      const bool masked = (log_seqs[b * LL + q] == 0) || (kk > q);
      s_sc[m * 256 + kk] = masked ? NEGV : sc;
    }
  }
  __syncthreads();

  // ---- softmax: row = lane&15, halves of 128 keys combined via shfl_xor(16)
  {
    const int r = lane & 15, hf = lane >> 4;
    const int base = r * 256 + hf * 128;
    float mx = -3.4e38f;
    for (int k = 0; k < 128; ++k) mx = fmaxf(mx, s_sc[base + k]);
    mx = fmaxf(mx, __shfl_xor(mx, 16, 32));
    float sum = 0.0f;
    for (int k = 0; k < 128; ++k) {
      const float e = __expf(s_sc[base + k] - mx);
      s_sc[base + k] = e;
      sum += e;
    }
    sum += __shfl_xor(sum, 16, 32);
    const float inv = 1.0f / sum;
    for (int k = 0; k < 128; ++k) p_h[base + k] = (_Float16)(s_sc[base + k] * inv);
  }
  __syncthreads();

  // ---- O = P (V + posV) via WMMA (two 16-wide d tiles) ----
  v8f o0 = {0.f, 0.f, 0.f, 0.f, 0.f, 0.f, 0.f, 0.f};
  v8f o1 = {0.f, 0.f, 0.f, 0.f, 0.f, 0.f, 0.f, 0.f};
#pragma unroll
  for (int kc = 0; kc < 8; ++kc) {
    const v16h pa = frag_ld_a_h(p_h + kc * 32, 256);
    const v16h b0 = frag_ld_b_h(vT + 0 * 256 + kc * 32, 256);
    const v16h b1 = frag_ld_b_h(vT + 16 * 256 + kc * 32, 256);
    o0 = wmma16(pa, b0, o0);
    o1 = wmma16(pa, b1, o1);
  }

  // ---- O += P . tV (q-dependent gather, VALU) + store ----
#pragma unroll
  for (int r = 0; r < 8; ++r) {
    const int m = mo + r;
    float acc0 = 0.0f, acc1 = 0.0f;
    const _Float16* pr = p_h + m * 256;
    const int* tr = tmr + m * 256;
    for (int k = 0; k < 256; ++k) {
      const float pw = (float)pr[k];
      const float* te = timeV + (size_t)tr[k] * DD + hd0;
      acc0 = fmaf(pw, te[n], acc0);
      acc1 = fmaf(pw, te[16 + n], acc1);
    }
    const size_t ob = ((size_t)(b * LL + q0 + m)) * DD + hd0;
    out[ob + n]      = o0[r] + acc0;
    out[ob + 16 + n] = o1[r] + acc1;
  }
}

// ---------------------------------------------------------------------------
// Final LN + pos/neg logits: one block (128 thr) per (b,l) row.
// ---------------------------------------------------------------------------

__global__ void final_kernel(const float* __restrict__ seqs,
                             const float* __restrict__ g,
                             const float* __restrict__ bta,
                             const float* __restrict__ item_emb,
                             const int* __restrict__ pos_seqs,
                             const int* __restrict__ neg_seqs,
                             float* __restrict__ out) {
  __shared__ float red[DD];
  const int row = blockIdx.x, t = threadIdx.x;
  const float v = seqs[(size_t)row * DD + t];
  red[t] = v;
  __syncthreads();
  for (int s = 64; s > 0; s >>= 1) { if (t < s) red[t] += red[t + s]; __syncthreads(); }
  const float mean = red[0] * (1.0f / DD);
  __syncthreads();
  const float d = v - mean;
  red[t] = d * d;
  __syncthreads();
  for (int s = 64; s > 0; s >>= 1) { if (t < s) red[t] += red[t + s]; __syncthreads(); }
  const float var = red[0] * (1.0f / DD);
  __syncthreads();
  const float f = d * rsqrtf(var + 1e-8f) * g[t] + bta[t];

  const int pid = pos_seqs[row];
  red[t] = f * item_emb[(size_t)pid * DD + t];
  __syncthreads();
  for (int s = 64; s > 0; s >>= 1) { if (t < s) red[t] += red[t + s]; __syncthreads(); }
  if (t == 0) out[row] = red[0];
  __syncthreads();

  const int nid = neg_seqs[row];
  red[t] = f * item_emb[(size_t)nid * DD + t];
  __syncthreads();
  for (int s = 64; s > 0; s >>= 1) { if (t < s) red[t] += red[t + s]; __syncthreads(); }
  if (t == 0) out[MM + row] = red[0];
}

// ---------------------------------------------------------------------------
// Launch
// ---------------------------------------------------------------------------

extern "C" void kernel_launch(void* const* d_in, const int* in_sizes, int n_in,
                              void* d_out, int out_size, void* d_ws, size_t ws_size,
                              hipStream_t stream) {
  (void)in_sizes; (void)n_in; (void)out_size; (void)ws_size;
  const int*   log_seqs = (const int*)d_in[1];
  const int*   tmx      = (const int*)d_in[2];
  const int*   pos_seqs = (const int*)d_in[3];
  const int*   neg_seqs = (const int*)d_in[4];
  const float* item_emb = (const float*)d_in[5];
  const float* posK     = (const float*)d_in[6];
  const float* posV     = (const float*)d_in[7];
  const float* timeK    = (const float*)d_in[8];
  const float* timeV    = (const float*)d_in[9];
  const float* ln1_g = (const float*)d_in[10];
  const float* ln1_b = (const float*)d_in[11];
  const float* Wq = (const float*)d_in[12];
  const float* bq = (const float*)d_in[13];
  const float* Wk = (const float*)d_in[14];
  const float* bk = (const float*)d_in[15];
  const float* Wv = (const float*)d_in[16];
  const float* bv = (const float*)d_in[17];
  const float* ln2_g = (const float*)d_in[18];
  const float* ln2_b = (const float*)d_in[19];
  const float* W1 = (const float*)d_in[20];
  const float* b1 = (const float*)d_in[21];
  const float* W2 = (const float*)d_in[22];
  const float* b2 = (const float*)d_in[23];
  const float* lnf_g = (const float*)d_in[24];
  const float* lnf_b = (const float*)d_in[25];

  float* ws = (float*)d_ws;
  const size_t SZ = (size_t)MM * DD;   // 262144 floats
  float* seqs = ws;
  float* q_in = ws + 1 * SZ;
  float* Qb   = ws + 2 * SZ;
  float* Kb   = ws + 3 * SZ;
  float* Vb   = ws + 4 * SZ;
  float* attn = ws + 5 * SZ;
  float* hbuf = ws + 6 * SZ;

  embed_kernel<<<MM, DD, 0, stream>>>(item_emb, log_seqs, seqs);

  for (int i = 0; i < 2; ++i) {
    const size_t wo = (size_t)i * DD * DD;
    const size_t bo = (size_t)i * DD;
    ln_kernel<<<MM, DD, 0, stream>>>(seqs, ln1_g + bo, ln1_b + bo, q_in);
    gemm128_kernel<false, false><<<MM / 32, 64, 0, stream>>>(q_in, Wq + wo, bq + bo, nullptr, Qb);
    gemm128_kernel<false, false><<<MM / 32, 64, 0, stream>>>(seqs, Wk + wo, bk + bo, nullptr, Kb);
    gemm128_kernel<false, false><<<MM / 32, 64, 0, stream>>>(seqs, Wv + wo, bv + bo, nullptr, Vb);
    attn_kernel<<<BB * HH * (LL / 16), 32, 0, stream>>>(
        Qb, Kb, Vb, posK, posV, timeK, timeV, tmx, log_seqs, attn);
    add_kernel<<<(MM * DD) / 256, 256, 0, stream>>>(q_in, attn, seqs, MM * DD);
    ln_kernel<<<MM, DD, 0, stream>>>(seqs, ln2_g + bo, ln2_b + bo, seqs);
    gemm128_kernel<true, false><<<MM / 32, 64, 0, stream>>>(seqs, W1 + wo, b1 + bo, nullptr, hbuf);
    gemm128_kernel<false, true><<<MM / 32, 64, 0, stream>>>(hbuf, W2 + wo, b2 + bo, seqs, seqs);
    mask_kernel<<<MM, DD, 0, stream>>>(seqs, log_seqs);
  }

  final_kernel<<<MM, DD, 0, stream>>>(seqs, lnf_g, lnf_b, item_emb,
                                      pos_seqs, neg_seqs, (float*)d_out);
}
